// GNNEncoder_53601191854606
// MI455X (gfx1250) — compile-verified, matrix operand order checked by reading, not run on hardware
//
#include <hip/hip_runtime.h>
#include <hip/hip_bf16.h>
#include <math.h>

typedef __attribute__((ext_vector_type(16))) __bf16 v16bf;
typedef __attribute__((ext_vector_type(8)))  float  v8f;

// ---------------------------------------------------------------------------
// WMMA GEMM: C[M x Nout] = A[M x K] * B[K x Nout]   (row-major, f32 in/out,
// bf16 multiply via v_wmma_f32_16x16x32_bf16, f32 accumulate).
// One wave computes one 16x16 tile of C; K templated so the K-loop fully
// unrolls into K/32 back-to-back WMMAs (4 for K=128). M % 16 == 0 (50000 ok),
// K % 32 == 0, Nout % 16 == 0 (128/64 ok).
// ---------------------------------------------------------------------------
template <int K>
__global__ void gcn_wmma_gemm(const float* __restrict__ A,
                              const float* __restrict__ B,
                              float* __restrict__ C,
                              int Nout, int tilesN, int totalTiles) {
    int wave = (int)((blockIdx.x * blockDim.x + threadIdx.x) >> 5);
    int lane = threadIdx.x & 31;
    if (wave >= totalTiles) return;            // wave-uniform exit: EXEC stays all-1s

    int tM   = wave / tilesN;
    int tN   = wave - tM * tilesN;
    int half = lane >> 4;                      // 0 or 1: selects K-base per ISA layout
    int rc   = lane & 15;                      // A row within tile / B col within tile

    const float* __restrict__ Arow = A + (size_t)(tM * 16 + rc) * K;
    int colB = tN * 16 + rc;

    v8f acc = {};
    #pragma unroll
    for (int k0 = 0; k0 < K; k0 += 32) {
        int kbase = k0 + 8 * half;
        v16bf a, b;
        // 16-bit A/B fragment element e -> K = kbase + e (e<8) or kbase + 8 + e (e>=8)
        #pragma unroll
        for (int e = 0; e < 16; ++e) {
            int k = kbase + e + ((e >= 8) ? 8 : 0);
            a[e] = (__bf16)Arow[k];
            b[e] = (__bf16)B[(size_t)k * Nout + colB];
        }
        acc = __builtin_amdgcn_wmma_f32_16x16x32_bf16(
            /*neg_a=*/false, a, /*neg_b=*/false, b,
            /*c_mod=*/(short)0, acc, /*reuse_a=*/false, /*reuse_b=*/false);
    }

    // C/D layout: VGPR r -> row tM*16 + r + 8*half, col tN*16 + rc
    int rowC = tM * 16 + 8 * half;
    int colC = tN * 16 + rc;
    #pragma unroll
    for (int r = 0; r < 8; ++r)
        C[(size_t)(rowC + r) * Nout + colC] = acc[r];
}

// ---------------------------------------------------------------------------
__global__ void zero_f32(float* __restrict__ p, int n) {
    int i = blockIdx.x * blockDim.x + threadIdx.x;
    if (i < n) p[i] = 0.0f;
}

__global__ void deg_accum(const int* __restrict__ dst, float* __restrict__ deg, int E) {
    int e = blockIdx.x * blockDim.x + threadIdx.x;
    if (e < E) atomicAdd(&deg[dst[e]], 1.0f);
}

__global__ void deg_to_dis(float* __restrict__ d, int n) {
    int i = blockIdx.x * blockDim.x + threadIdx.x;
    if (i < n) d[i] = rsqrtf(1.0f + d[i]);     // self-loop included in degree
}

// One thread per (edge, 4-feature group); with F=128, one wave == one edge,
// lanes cover the full feature row -> coalesced 512B gather + contiguous
// atomics that resolve in the 192MB L2 (h/agg are only 25.6MB each).
__global__ void edge_scatter(const int* __restrict__ src, const int* __restrict__ dst,
                             const float* __restrict__ dis, const float* __restrict__ h,
                             float* __restrict__ agg, int E, int F) {
    int fg    = F >> 2;
    int idx   = blockIdx.x * blockDim.x + threadIdx.x;
    int total = E * fg;
    if (idx >= total) return;
    int e = idx / fg;
    int g = idx - e * fg;
    int s = src[e], d = dst[e];
    float c = dis[s] * dis[d];
    float4 hv = ((const float4*)(h + (size_t)s * F))[g];
    float* ap = agg + (size_t)d * F + (g << 2);
    atomicAdd(ap + 0, c * hv.x);
    atomicAdd(ap + 1, c * hv.y);
    atomicAdd(ap + 2, c * hv.z);
    atomicAdd(ap + 3, c * hv.w);
}

// z = agg + dis^2*h + bias (in place on agg) and accumulate per-feature sum/sumsq.
// blockDim.x == F; each block handles `rows` rows -> 2 atomics per feature/block.
__global__ void self_bias_stats(const float* __restrict__ h, const float* __restrict__ dis,
                                const float* __restrict__ bias, float* __restrict__ z,
                                float* __restrict__ sum, float* __restrict__ sumsq,
                                int N, int F, int rows) {
    int f  = threadIdx.x;
    int r0 = blockIdx.x * rows;
    int r1 = r0 + rows; if (r1 > N) r1 = N;
    float b = bias[f], s = 0.0f, s2 = 0.0f;
    for (int r = r0; r < r1; ++r) {
        float di  = dis[r];
        size_t o  = (size_t)r * F + f;
        float val = z[o] + di * di * h[o] + b;
        z[o] = val;
        s  += val;
        s2 += val * val;
    }
    atomicAdd(&sum[f], s);
    atomicAdd(&sumsq[f], s2);
}

__global__ void bn_finalize(const float* __restrict__ sum, const float* __restrict__ sumsq,
                            const float* __restrict__ g, const float* __restrict__ be,
                            float* __restrict__ scale, float* __restrict__ shift,
                            int N, float eps) {
    int f = threadIdx.x;
    float m   = sum[f] / (float)N;
    float var = sumsq[f] / (float)N - m * m;   // biased variance, matches reference
    float inv = rsqrtf(var + eps);
    float sc  = g[f] * inv;
    scale[f] = sc;
    shift[f] = be[f] - m * sc;
}

__global__ void bn_elu(const float* __restrict__ z, const float* __restrict__ scale,
                       const float* __restrict__ shift, float* __restrict__ out,
                       int total, int fmask) {
    int i = blockIdx.x * blockDim.x + threadIdx.x;
    if (i >= total) return;
    int f = i & fmask;                          // F is a power of two (128/64)
    float v = scale[f] * z[i] + shift[f];
    out[i] = v > 0.0f ? v : expm1f(v);
}

// ---------------------------------------------------------------------------
extern "C" void kernel_launch(void* const* d_in, const int* in_sizes, int n_in,
                              void* d_out, int out_size, void* d_ws, size_t ws_size,
                              hipStream_t stream) {
    const int D_IN = 128;
    const int N = in_sizes[0] / D_IN;
    const int E = in_sizes[1] / 2;

    const float* x   = (const float*)d_in[0];
    const int*   ei  = (const int*)d_in[1];
    const int*   src = ei;
    const int*   dst = ei + E;
    const float* W[3]  = {(const float*)d_in[2], (const float*)d_in[6],  (const float*)d_in[10]};
    const float* bb[3] = {(const float*)d_in[3], (const float*)d_in[7],  (const float*)d_in[11]};
    const float* gg[3] = {(const float*)d_in[4], (const float*)d_in[8],  (const float*)d_in[12]};
    const float* be[3] = {(const float*)d_in[5], (const float*)d_in[9],  (const float*)d_in[13]};

    float* ws    = (float*)d_ws;
    float* dis   = ws;                          // N
    float* act   = dis + N;                     // N*128
    float* h     = act + (size_t)N * 128;       // N*128
    float* agg   = h   + (size_t)N * 128;       // N*128
    float* sum   = agg + (size_t)N * 128;       // 128
    float* sumsq = sum + 128;                   // 128
    float* scale = sumsq + 128;                 // 128
    float* shift = scale + 128;                 // 128

    // ---- degree -> dis = rsqrt(1 + indeg) ----
    zero_f32<<<(N + 255) / 256, 256, 0, stream>>>(dis, N);
    deg_accum<<<(E + 255) / 256, 256, 0, stream>>>(dst, dis, E);
    deg_to_dis<<<(N + 255) / 256, 256, 0, stream>>>(dis, N);

    const int Fout[3] = {128, 128, 64};
    const float* inp[3] = {x, act, act};
    float* outp[3] = {act, act, (float*)d_out};

    for (int l = 0; l < 3; ++l) {
        int F = Fout[l];

        // GEMM: h = inp @ W  (one wave per 16x16 tile, 8 waves per block, K=128 unrolled)
        int tilesN = F / 16;
        int totalTiles = (N / 16) * tilesN;
        int gemmBlocks = (totalTiles + 7) / 8;
        gcn_wmma_gemm<128><<<gemmBlocks, 256, 0, stream>>>(inp[l], W[l], h, F, tilesN, totalTiles);

        // zero aggregation buffer + stats
        int nAgg = N * F;
        zero_f32<<<(nAgg + 255) / 256, 256, 0, stream>>>(agg, nAgg);
        zero_f32<<<1, 256, 0, stream>>>(sum, 256);   // sum[128] + sumsq[128] contiguous

        // edge scatter: agg[dst] += dis[src]*dis[dst] * h[src]
        int total = E * (F >> 2);
        edge_scatter<<<(total + 255) / 256, 256, 0, stream>>>(src, dst, dis, h, agg, E, F);

        // self-loop + bias + BN batch statistics
        const int ROWS = 64;
        self_bias_stats<<<(N + ROWS - 1) / ROWS, F, 0, stream>>>(h, dis, bb[l], agg, sum, sumsq, N, F, ROWS);

        // BN scale/shift, then BN + ELU
        bn_finalize<<<1, F, 0, stream>>>(sum, sumsq, gg[l], be[l], scale, shift, N, 1e-5f);
        bn_elu<<<(nAgg + 255) / 256, 256, 0, stream>>>(agg, scale, shift, outp[l], nAgg, F - 1);
    }
}
